// Block_21809843929850
// MI455X (gfx1250) — matrix operand order repaired and checked by reading, not compiled, not run on hardware
//
#include <hip/hip_runtime.h>

// CDNA5 / gfx1250 fused MoD-transformer block.
// Dead attention branch (overwritten scatter) is skipped; MLP runs on bf16 WMMA
// with double-buffered async-to-LDS staging (ASYNCcnt pipeline). Wave tile is
// 32Mx64N so each B fragment feeds 2 WMMAs and each A fragment 4 (LDS-BW relief).

typedef __attribute__((ext_vector_type(16))) __bf16   v16bf;
typedef __attribute__((ext_vector_type(8)))  float    v8f;
typedef __attribute__((ext_vector_type(4)))  unsigned u32x4;

#define DIM   1024
#define SEQ   2048
#define BATCH 8
#define NROWS (BATCH * SEQ)   // 16384
#define HID   4096
#define KSEL  512

// ---------------- CDNA5 async-copy primitives (inline asm; portable across toolchains) ----
__device__ __forceinline__ void async_b128(const void* gaddr, unsigned lds_off) {
  // per-lane: LDS[lds_off] = MEM[gaddr], 16B; tracked by ASYNCcnt
  asm volatile("global_load_async_to_lds_b128 %0, %1, off"
               :: "v"(lds_off), "v"(gaddr) : "memory");
}
__device__ __forceinline__ void wait_async_10() { asm volatile("s_wait_asynccnt 10" ::: "memory"); }
__device__ __forceinline__ void wait_async_0()  { asm volatile("s_wait_asynccnt 0"  ::: "memory"); }

// ---------------- fast branch-free tanh-GELU:  0.5*(1+tanh(y)) == 1/(1+exp(-2y)) ----------
__device__ __forceinline__ float gelu_f(float v) {
  float y = 0.7978845608f * (v + 0.044715f * v * v * v);
  float t = -2.8853900818f * y;          // -2y * log2(e)
#if __has_builtin(__builtin_amdgcn_exp2f)
  float e = __builtin_amdgcn_exp2f(t);   // v_exp_f32
#else
  float e = __builtin_exp2f(t);
#endif
#if __has_builtin(__builtin_amdgcn_rcpf)
  return v * __builtin_amdgcn_rcpf(1.f + e);  // v_rcp_f32
#else
  return v / (1.f + e);
#endif
}

// ---------------- kernel 1: router scores (one wave32 per token) ----------------
__global__ void __launch_bounds__(256) router_kernel(
    const float* __restrict__ x, const float* __restrict__ rw,
    const float* __restrict__ rb, float* __restrict__ rscore)
{
  int wave = threadIdx.x >> 5, lane = threadIdx.x & 31;
  int row  = blockIdx.x * 8 + wave;
  const float4* xr = (const float4*)(x + (size_t)row * DIM);
  const float4* wr = (const float4*)rw;
  float s = 0.f;
#pragma unroll
  for (int c = 0; c < (DIM / 4) / 32; ++c) {
    float4 a = xr[c * 32 + lane];
    float4 b = wr[c * 32 + lane];
    s += a.x * b.x + a.y * b.y + a.z * b.z + a.w * b.w;
  }
#pragma unroll
  for (int off = 16; off; off >>= 1) s += __shfl_down(s, off, 32);
  if (lane == 0) rscore[row] = s + rb[0];
}

// ---------------- kernel 2: top-k selection mask via MSB radix-select ----------------
__global__ void __launch_bounds__(256) topk_mask_kernel(
    const float* __restrict__ rscore, int* __restrict__ mask)
{
  __shared__ unsigned key[SEQ];
  __shared__ int cnt;
  int b = blockIdx.x, tid = threadIdx.x;
  for (int i = tid; i < SEQ; i += 256) {
    unsigned u = __float_as_uint(rscore[b * SEQ + i]);
    key[i] = (u & 0x80000000u) ? ~u : (u | 0x80000000u);  // order-preserving map
  }
  __syncthreads();
  unsigned prefix = 0; int kk = KSEL;
  for (int bit = 31; bit >= 0; --bit) {
    if (tid == 0) cnt = 0;
    __syncthreads();
    unsigned hi = (bit == 31) ? 0u : ~((2u << bit) - 1u);
    int local = 0;
    for (int i = tid; i < SEQ; i += 256) {
      unsigned u = key[i];
      if ((u & hi) == prefix && ((u >> bit) & 1u)) local++;
    }
    atomicAdd(&cnt, local);
    __syncthreads();
    int c = cnt;
    if (c >= kk) prefix |= (1u << bit); else kk -= c;
    __syncthreads();
  }
  unsigned kth = prefix;
  if (tid == 0) cnt = 0;
  __syncthreads();
  int local = 0;
  for (int i = tid; i < SEQ; i += 256) if (key[i] > kth) local++;
  atomicAdd(&cnt, local);
  __syncthreads();
  int g = cnt;
  for (int i = tid; i < SEQ; i += 256) mask[b * SEQ + i] = (key[i] > kth) ? 1 : 0;
  __syncthreads();
  if (tid == 0) {  // ties broken by lowest index (jax top_k convention)
    int need = KSEL - g;
    for (int i = 0; i < SEQ && need > 0; ++i)
      if (key[i] == kth) { mask[b * SEQ + i] = 1; need--; }
  }
}

// ---------------- kernel 3: x_new = x + (mask ? LN1(x[reversed row]) : 0) ----------------
__global__ void __launch_bounds__(256) scatter_residual_kernel(
    const float* __restrict__ x, const int* __restrict__ mask,
    const float* __restrict__ g, const float* __restrict__ bb,
    float* __restrict__ out)
{
  __shared__ float red[256], red2[256];
  int t = blockIdx.x, tid = threadIdx.x;
  int b = t >> 11, i = t & (SEQ - 1);
  const float* xr = x + (size_t)t * DIM;
  float* orow = out + (size_t)t * DIM;
  if (!mask[t]) {
    for (int j = tid; j < DIM; j += 256) orow[j] = xr[j];
    return;
  }
  const float* sr = x + (size_t)(b * SEQ + (SEQ - 1 - i)) * DIM;
  float s = 0.f, s2 = 0.f;
  for (int j = tid; j < DIM; j += 256) { float v = sr[j]; s += v; s2 += v * v; }
  red[tid] = s; red2[tid] = s2; __syncthreads();
  for (int off = 128; off; off >>= 1) {
    if (tid < off) { red[tid] += red[tid + off]; red2[tid] += red2[tid + off]; }
    __syncthreads();
  }
  float mean = red[0] * (1.f / DIM);
  float rstd = rsqrtf(red2[0] * (1.f / DIM) - mean * mean + 1e-5f);
  for (int j = tid; j < DIM; j += 256)
    orow[j] = xr[j] + ((sr[j] - mean) * rstd * g[j] + bb[j]);
}

// ---------------- kernel 4: h2 = bf16(LN2(x_new)) ----------------
__global__ void __launch_bounds__(256) ln_bf16_kernel(
    const float* __restrict__ xn, const float* __restrict__ g,
    const float* __restrict__ bb, __bf16* __restrict__ h2)
{
  __shared__ float red[256], red2[256];
  int t = blockIdx.x, tid = threadIdx.x;
  const float* xr = xn + (size_t)t * DIM;
  float s = 0.f, s2 = 0.f;
  for (int j = tid; j < DIM; j += 256) { float v = xr[j]; s += v; s2 += v * v; }
  red[tid] = s; red2[tid] = s2; __syncthreads();
  for (int off = 128; off; off >>= 1) {
    if (tid < off) { red[tid] += red[tid + off]; red2[tid] += red2[tid + off]; }
    __syncthreads();
  }
  float mean = red[0] * (1.f / DIM);
  float rstd = rsqrtf(red2[0] * (1.f / DIM) - mean * mean + 1e-5f);
  for (int j = tid; j < DIM; j += 256)
    h2[(size_t)t * DIM + j] = (__bf16)((xr[j] - mean) * rstd * g[j] + bb[j]);
}

// ---------------- kernel 5: f32 -> bf16 weight downcast ----------------
__global__ void __launch_bounds__(256) f32_to_bf16_kernel(
    const float* __restrict__ src, __bf16* __restrict__ dst, int n)
{
  int i = blockIdx.x * 256 + threadIdx.x;
  if (i < n) dst[i] = (__bf16)src[i];
}

// ---------------- WMMA fragment loaders (16-bit A 16x32 / B 32x16 ISA layouts) ----------------
// A: lanes 0-15 hold row M=lane, K = {k..k+7, k+16..k+23}; lanes 16-31: +8 on each group.
// B: lanes 0-15 hold col N=lane, K = {k..k+7, k+16..k+23}; lanes 16-31: +8 on each group.
__device__ __forceinline__ v16bf load_a16(const __bf16* p) {
  union { u32x4 u[2]; v16bf v; } f;
  f.u[0] = *(const u32x4*)(p);
  f.u[1] = *(const u32x4*)(p + 16);
  return f.v;
}
__device__ __forceinline__ v16bf load_b16(const __bf16* p) {
  union { u32x4 u[2]; v16bf v; } f;
  f.u[0] = *(const u32x4*)(p);
  f.u[1] = *(const u32x4*)(p + 16);
  return f.v;
}

// ---------------- WMMA GEMM with double-buffered async-to-LDS staging ----------------
// block = 256 threads = 8 waves; block tile 64(M) x 256(N); K chunk 64.
// wave tile 32(M) x 64(N): B fragments reused 2x, A fragments 4x.
// DOGELU=1: C(bf16) = gelu(A @ W^T + bias)   DOGELU=0: outf(f32) += A @ W^T + bias
#define BKPAD 80   // 64 k-elements padded to 160B rows (keeps 16B ds_load_b128 alignment)

template<int KDIM, bool DOGELU>
__global__ void __launch_bounds__(256) gemm_lds_kernel(
    const __bf16* __restrict__ A,   // [M, KDIM]
    const __bf16* __restrict__ W,   // [N, KDIM]
    const float* __restrict__ bias, // [N]
    __bf16* __restrict__ Cb,        // DOGELU: [M, Nld]
    float* __restrict__ Of,         // !DOGELU: [M, Nld] accumulated
    int Nld)                        // leading dim of output
{
  __shared__ __bf16 Bs[2][256][BKPAD];  // 80 KiB
  __shared__ __bf16 As[2][64][BKPAD];   // 20 KiB
  const int tid  = threadIdx.x;
  const int wave = tid >> 5, lane = tid & 31;
  const int mblock = blockIdx.y * 64;
  const int nblock = blockIdx.x * 256;
  const int akb = (lane < 16) ? 0 : 8;
  const int bkb = (lane < 16) ? 0 : 8;
  const int arow0 = (wave >> 2) * 32 + (lane & 15);   // wave's first 16-row A group
  const int brow0 = (wave & 3) * 64 + (lane & 15);

  auto issue_stage = [&](int s, int k0) {
    // B tile: 256 rows x 64 k (8 x 16B chunks per row); thread -> row tid
    const __bf16* gb = W + (size_t)(nblock + tid) * KDIM + k0;
    unsigned lb = (unsigned)(size_t)&Bs[s][tid][0];
#pragma unroll
    for (int c = 0; c < 8; ++c) async_b128(gb + c * 8, lb + c * 16u);
    // A tile: 64 rows x 64 k; thread -> (row tid&63, chunks (tid>>6), (tid>>6)+4)
    const int arw = tid & 63, ach = tid >> 6;
    const __bf16* ga = A + (size_t)(mblock + arw) * KDIM + k0;
    unsigned la = (unsigned)(size_t)&As[s][arw][0];
    async_b128(ga + ach * 8,       la + ach * 16u);
    async_b128(ga + (ach + 4) * 8, la + (ach + 4) * 16u);
    __builtin_prefetch(gb + 128, 0, 1);  // warm GL2 two stages ahead
  };

  v8f acc[2][4];
#pragma unroll
  for (int mt = 0; mt < 2; ++mt)
#pragma unroll
    for (int t = 0; t < 4; ++t) acc[mt][t] = v8f{};

  issue_stage(0, 0);
  int s = 0;
  for (int k0 = 0; k0 < KDIM; k0 += 64, s ^= 1) {
    const bool more = (k0 + 64) < KDIM;
    if (more) { issue_stage(s ^ 1, k0 + 64); wait_async_10(); } // stage s complete (in-order)
    else      { wait_async_0(); }
    __syncthreads();                       // stage s visible to all waves
#pragma unroll
    for (int ks = 0; ks < 64; ks += 32) {
      v16bf a0 = load_a16(&As[s][arow0][ks + akb]);
      v16bf a1 = load_a16(&As[s][arow0 + 16][ks + akb]);
#pragma unroll
      for (int t = 0; t < 4; ++t) {
        v16bf b = load_b16(&Bs[s][brow0 + 16 * t][ks + bkb]);
        acc[0][t] = __builtin_amdgcn_wmma_f32_16x16x32_bf16(
            false, a0, false, b, (short)0, acc[0][t], false, false);
        acc[1][t] = __builtin_amdgcn_wmma_f32_16x16x32_bf16(
            false, a1, false, b, (short)0, acc[1][t], false, false);
      }
    }
    __syncthreads();                       // all waves done reading before next overwrite
  }

  // C layout: VGPR j -> (M = mbase + j + (lane<16?0:8), N = nbase + 16*t + (lane&15))
  const int c0 = nblock + (wave & 3) * 64 + (lane & 15);
#pragma unroll
  for (int mt = 0; mt < 2; ++mt) {
    const int r0 = mblock + (wave >> 2) * 32 + mt * 16 + ((lane < 16) ? 0 : 8);
#pragma unroll
    for (int t = 0; t < 4; ++t) {
      int c = c0 + t * 16;
      float bv = bias[c];
#pragma unroll
      for (int j = 0; j < 8; ++j) {
        float v = acc[mt][t][j] + bv;
        if constexpr (DOGELU) Cb[(size_t)(r0 + j) * Nld + c] = (__bf16)gelu_f(v);
        else                  Of[(size_t)(r0 + j) * Nld + c] += v;
      }
    }
  }
}

// ---------------- host launcher ----------------
extern "C" void kernel_launch(void* const* d_in, const int* in_sizes, int n_in,
                              void* d_out, int out_size, void* d_ws, size_t ws_size,
                              hipStream_t stream) {
  (void)in_sizes; (void)n_in; (void)out_size; (void)ws_size;
  const float* x        = (const float*)d_in[0];
  const float* router_w = (const float*)d_in[1];
  const float* router_b = (const float*)d_in[2];
  const float* ln1_g    = (const float*)d_in[3];
  const float* ln1_b    = (const float*)d_in[4];
  // d_in[5..7] = wq, wk, wv: feed a dead (overwritten) scatter in the reference — skipped.
  const float* ln2_g    = (const float*)d_in[8];
  const float* ln2_b    = (const float*)d_in[9];
  const float* fc_w     = (const float*)d_in[10];
  const float* fc_b     = (const float*)d_in[11];
  const float* proj_w   = (const float*)d_in[12];
  const float* proj_b   = (const float*)d_in[13];
  float* out = (float*)d_out;

  char* ws = (char*)d_ws;
  float*  rscore = (float*)ws;                 ws += (size_t)NROWS * 4;
  int*    mask   = (int*)ws;                   ws += (size_t)NROWS * 4;
  __bf16* h2     = (__bf16*)ws;                ws += (size_t)NROWS * DIM * 2;
  __bf16* fcw    = (__bf16*)ws;                ws += (size_t)HID * DIM * 2;
  __bf16* pjw    = (__bf16*)ws;                ws += (size_t)DIM * HID * 2;
  __bf16* act    = (__bf16*)ws;                // NROWS * HID bf16

  router_kernel<<<NROWS / 8, 256, 0, stream>>>(x, router_w, router_b, rscore);
  topk_mask_kernel<<<BATCH, 256, 0, stream>>>(rscore, mask);
  scatter_residual_kernel<<<NROWS, 256, 0, stream>>>(x, mask, ln1_g, ln1_b, out);
  ln_bf16_kernel<<<NROWS, 256, 0, stream>>>(out, ln2_g, ln2_b, h2);
  f32_to_bf16_kernel<<<(HID * DIM) / 256, 256, 0, stream>>>(fc_w, fcw, HID * DIM);
  f32_to_bf16_kernel<<<(DIM * HID) / 256, 256, 0, stream>>>(proj_w, pjw, DIM * HID);
  gemm_lds_kernel<DIM, true><<<dim3(HID / 256, NROWS / 64), 256, 0, stream>>>(
      h2, fcw, fc_b, act, nullptr, HID);
  gemm_lds_kernel<HID, false><<<dim3(DIM / 256, NROWS / 64), 256, 0, stream>>>(
      act, pjw, proj_b, nullptr, out, DIM);
}
